// APN_50414326121132
// MI455X (gfx1250) — matrix operand order repaired and checked by reading, not compiled, not run on hardware
//
#include <hip/hip_runtime.h>
#include <hip/hip_bf16.h>
#include <math.h>

typedef __attribute__((ext_vector_type(16))) _Float16 v16h;
typedef __attribute__((ext_vector_type(4)))  _Float16 v4h;
typedef __attribute__((ext_vector_type(8)))  float    v8f;

#define N_OBJ  50000
#define E_PER  266667

// Flattened input order: jax pytree flattening of setup_inputs() dict
// (dict keys sorted alphabetically, DFS; lin{w,b} -> b,w; layers list in order).
enum {
  I_DST_G = 0, I_DST_GOAL, I_FEAT_G, I_FEAT_GOAL, I_OBJECT_VEC,
  I_EMB1_B, I_EMB1_W, I_EMB2_B, I_EMB2_W,
  I_FC1_B, I_FC1_W, I_FC2_B, I_FC2_W, I_FC3_B, I_FC3_W,
  I_L0_W, I_L0_WSELF, I_L0_B, I_L0_WGD, I_L0_WGS,
  I_L1_W, I_L1_WSELF, I_L1_B, I_L1_WGD, I_L1_WGS,
  I_M1_B, I_M1_W, I_M2_B, I_M2_W,
  I_P1_B, I_P1_W, I_P2_B, I_P2_W, I_P3_B, I_P3_W,
  I_PRELU_A,
  I_Q1O_B, I_Q1O_W, I_Q1S_B, I_Q1S_W,
  I_Q2O_B, I_Q2O_W, I_Q2S_B, I_Q2S_W,
  I_Q3O_B, I_Q3O_W, I_Q3S_B, I_Q3S_W,
  I_SC1_B, I_SC1_W, I_SC2_B, I_SC2_W,
  I_SRC_G, I_SRC_GOAL
};

__device__ __forceinline__ float sigf(float x) { return 1.f / (1.f + __expf(-x)); }

template <int ACT>
__device__ __forceinline__ float act_apply(float v, float a) {
  if (ACT == 1) return v > 0.f ? v : 0.f;
  if (ACT == 2) return v >= 0.f ? v : a * v;
  if (ACT == 3) return sigf(v);
  return v;
}

// LDS B-tile: 64 columns x 32 k, stored transposed [col][k] with stride 36
// halfs (72 B): 8-byte aligned for ds_load_b64, odd-ish stride vs banks.
#define BSTRIDE 36

// ---------------------------------------------------------------------------
// WMMA GEMM: out[M,N] = act( A[M,K] @ W[K,N] + bias + rowScalar[r]*extraVec[c] )
// MODE 0: store to out (ldc).  MODE 1: atomicAdd(out[col], (acc+bias)*rowW[row]).
// Block = 128 threads = 4 waves; block computes a 64(M) x 64(N) tile.
// B is staged f32->f16 through LDS once per block per 32-wide K chunk; each
// wave runs four v_wmma_f32_16x16x32_f16 per chunk. A rows are clamped to
// M-1 (no per-lane guards in the hot loop); stores are masked instead.
// ---------------------------------------------------------------------------
template <int ACT, int MODE>
__global__ __launch_bounds__(128)
void gemm_wmma_kernel(const float* __restrict__ A,
                      int M, int K, int lda,
                      const float* __restrict__ W, int N,
                      const float* __restrict__ bias,
                      const float* __restrict__ rowScalar,
                      const float* __restrict__ extraVec,
                      const float* __restrict__ rowW,
                      float* __restrict__ out, int ldc,
                      const float* __restrict__ prelu_a)
{
  __shared__ _Float16 sB[64 * BSTRIDE];

  const int lane    = threadIdx.x & 31;
  const int wave    = threadIdx.x >> 5;
  const int tileN0  = blockIdx.x * 64;
  const int rowBase = (blockIdx.y * 4 + wave) * 16;

  const int  arow   = rowBase + (lane & 15);
  const int  arowc  = arow < M ? arow : (M - 1);   // clamp: loads always legal
  const int  kbaseA = (lane >> 4) * 8;             // 0 or 8
  const int  kbaseB = (lane >> 4) * 8;             // 0 or 8
  const int  colIn  = lane & 15;
  const long aoff   = (long)arowc * lda;

  // cooperative B-fill mapping: 4 consecutive cols per thread, 4 k's
  const int fc4 = (threadIdx.x & 15) * 4;          // column group 0..60
  const int fkq = threadIdx.x >> 4;                // k quarter 0..7
  const float* Wb = W + tileN0 + fc4;

  v8f acc[4];
#pragma unroll
  for (int t = 0; t < 4; ++t)
#pragma unroll
    for (int r = 0; r < 8; ++r) acc[t][r] = 0.f;

  const int Kfull = K & ~31;

  for (int k0 = 0; k0 < Kfull; k0 += 32) {
    __syncthreads();                               // WAR vs previous reads
#pragma unroll
    for (int i = 0; i < 4; ++i) {
      const int k = fkq + i * 8;
      const float4 w4 = *(const float4*)(Wb + (long)(k0 + k) * N);
      sB[(fc4 + 0) * BSTRIDE + k] = (_Float16)w4.x;
      sB[(fc4 + 1) * BSTRIDE + k] = (_Float16)w4.y;
      sB[(fc4 + 2) * BSTRIDE + k] = (_Float16)w4.z;
      sB[(fc4 + 3) * BSTRIDE + k] = (_Float16)w4.w;
    }
    __syncthreads();

    // A fragment: two 4-float pairs at kbaseA and 16+kbaseA (unguarded)
    const float* pa0 = A + aoff + k0 + kbaseA;
    const float4 x0 = *(const float4*)(pa0);
    const float4 x1 = *(const float4*)(pa0 + 4);
    const float4 x2 = *(const float4*)(pa0 + 16);
    const float4 x3 = *(const float4*)(pa0 + 20);
    v16h av;
    av[0]=(_Float16)x0.x;  av[1]=(_Float16)x0.y;  av[2]=(_Float16)x0.z;  av[3]=(_Float16)x0.w;
    av[4]=(_Float16)x1.x;  av[5]=(_Float16)x1.y;  av[6]=(_Float16)x1.z;  av[7]=(_Float16)x1.w;
    av[8]=(_Float16)x2.x;  av[9]=(_Float16)x2.y;  av[10]=(_Float16)x2.z; av[11]=(_Float16)x2.w;
    av[12]=(_Float16)x3.x; av[13]=(_Float16)x3.y; av[14]=(_Float16)x3.z; av[15]=(_Float16)x3.w;

#pragma unroll
    for (int t = 0; t < 4; ++t) {
      const _Float16* bp = &sB[(colIn + t * 16) * BSTRIDE + kbaseB];
      const v4h b0 = *(const v4h*)(bp);
      const v4h b1 = *(const v4h*)(bp + 4);
      const v4h b2 = *(const v4h*)(bp + 16);
      const v4h b3 = *(const v4h*)(bp + 20);
      v16h bv;
#pragma unroll
      for (int h = 0; h < 4; ++h) {
        bv[h] = b0[h]; bv[4 + h] = b1[h]; bv[8 + h] = b2[h]; bv[12 + h] = b3[h];
      }
      acc[t] = __builtin_amdgcn_wmma_f32_16x16x32_f16(
          false, av, false, bv, (short)0, acc[t], false, false);
    }
  }

  if (K & 31) {                                    // single guarded tail chunk
    const int k0 = Kfull;
    __syncthreads();
#pragma unroll
    for (int i = 0; i < 4; ++i) {
      const int k = fkq + i * 8;
      float4 w4 = make_float4(0.f, 0.f, 0.f, 0.f);
      if (k0 + k < K) w4 = *(const float4*)(Wb + (long)(k0 + k) * N);
      sB[(fc4 + 0) * BSTRIDE + k] = (_Float16)w4.x;
      sB[(fc4 + 1) * BSTRIDE + k] = (_Float16)w4.y;
      sB[(fc4 + 2) * BSTRIDE + k] = (_Float16)w4.z;
      sB[(fc4 + 3) * BSTRIDE + k] = (_Float16)w4.w;
    }
    __syncthreads();

    v16h av;
#pragma unroll
    for (int j = 0; j < 16; ++j) {
      const int k = k0 + kbaseA + j + ((j < 8) ? 0 : 8);
      float v = 0.f;
      if (k < K) v = A[aoff + k];
      av[j] = (_Float16)v;
    }
#pragma unroll
    for (int t = 0; t < 4; ++t) {
      const _Float16* bp = &sB[(colIn + t * 16) * BSTRIDE + kbaseB];
      const v4h b0 = *(const v4h*)(bp);
      const v4h b1 = *(const v4h*)(bp + 4);
      const v4h b2 = *(const v4h*)(bp + 16);
      const v4h b3 = *(const v4h*)(bp + 20);
      v16h bv;
#pragma unroll
      for (int h = 0; h < 4; ++h) {
        bv[h] = b0[h]; bv[4 + h] = b1[h]; bv[8 + h] = b2[h]; bv[12 + h] = b3[h];
      }
      acc[t] = __builtin_amdgcn_wmma_f32_16x16x32_f16(
          false, av, false, bv, (short)0, acc[t], false, false);
    }
  }

  const float a = (ACT == 2 && prelu_a) ? prelu_a[0] : 0.f;

  if (MODE == 0) {
#pragma unroll
    for (int t = 0; t < 4; ++t) {
      const int col = tileN0 + t * 16 + colIn;
#pragma unroll
      for (int r = 0; r < 8; ++r) {
        const int row = rowBase + r + (lane >> 4) * 8;   // C/D layout
        if (row < M) {
          float v = acc[t][r];
          if (bias)      v += bias[col];
          if (rowScalar) v += rowScalar[row] * extraVec[col];
          out[(long)row * ldc + col] = act_apply<ACT>(v, a);
        }
      }
    }
  } else {
#pragma unroll
    for (int t = 0; t < 4; ++t) {
      const int col = tileN0 + t * 16 + colIn;
      float part = 0.f;
#pragma unroll
      for (int r = 0; r < 8; ++r) {
        const int row = rowBase + r + (lane >> 4) * 8;
        const float rw = (row < M) ? rowW[row] : 0.f;
        part += (acc[t][r] + (bias ? bias[col] : 0.f)) * rw;
      }
      part += __shfl_down(part, 16);
      if (lane < 16) atomicAdd(&out[col], part);
    }
  }
}

// in-place |a - b| for the scene input (keeps fabs out of the WMMA kernel;
// the packed-fadd+fabs fold crashes gfx1250 ISel there)
__global__ void absdiff_kernel(float* __restrict__ a, const float* __restrict__ b, long n)
{
  const long i = (long)blockIdx.x * blockDim.x + threadIdx.x;
  if (i < n) a[i] = fabsf(a[i] - b[i]);
}

// gs/gd gate logits: GS[n,e] = h[n,:] . wg_s[e,:]  (wave per node)
__global__ __launch_bounds__(256)
void gates_kernel(const float* __restrict__ h, int fi,
                  const float* __restrict__ wgs, const float* __restrict__ wgd,
                  float* __restrict__ GS, float* __restrict__ GD, int M)
{
  const int node = blockIdx.x * (blockDim.x >> 5) + (threadIdx.x >> 5);
  const int lane = threadIdx.x & 31;
  if (node >= M) return;
  const float* hr = h + (long)node * fi;
  float s0=0,s1=0,s2=0,d0=0,d1=0,d2=0;
  for (int k = lane; k < fi; k += 32) {
    const float hv = hr[k];
    s0 += hv * wgs[k]; s1 += hv * wgs[fi + k]; s2 += hv * wgs[2*fi + k];
    d0 += hv * wgd[k]; d1 += hv * wgd[fi + k]; d2 += hv * wgd[2*fi + k];
  }
  for (int off = 16; off > 0; off >>= 1) {
    s0 += __shfl_down(s0, off); s1 += __shfl_down(s1, off); s2 += __shfl_down(s2, off);
    d0 += __shfl_down(d0, off); d1 += __shfl_down(d1, off); d2 += __shfl_down(d2, off);
  }
  if (lane == 0) {
    GS[node*4+0]=s0; GS[node*4+1]=s1; GS[node*4+2]=s2;
    GD[node*4+0]=d0; GD[node*4+1]=d1; GD[node*4+2]=d2;
  }
}

// edge scatter: AGG[dst,:] += hW[src,:] * sigmoid(gs[src]+gd[dst])  (wave per edge)
__global__ __launch_bounds__(256)
void scatter_kernel(const float* __restrict__ HW, const float* __restrict__ GS,
                    const float* __restrict__ GD, int e,
                    const int* __restrict__ src, const int* __restrict__ dst,
                    int E, float* __restrict__ AGG)
{
  const int edge = blockIdx.x * (blockDim.x >> 5) + (threadIdx.x >> 5);
  const int lane = threadIdx.x & 31;
  if (edge >= E) return;
  const int s = src[edge], d = dst[edge];
  const float g = sigf(GS[s*4+e] + GD[d*4+e]);
  const float* hw = HW + (long)s * 256;
  float* ag = AGG + (long)d * 256;
#pragma unroll
  for (int j = 0; j < 8; ++j) {
    const int c = lane + j * 32;
    atomicAdd(&ag[c], hw[c] * g);
  }
}

__global__ void relu_copy_kernel(const float* __restrict__ src, float* __restrict__ dst,
                                 int M, int C, int lds, int ldd)
{
  const long i = (long)blockIdx.x * blockDim.x + threadIdx.x;
  if (i >= (long)M * C) return;
  const int row = (int)(i / C), col = (int)(i % C);
  const float v = src[(long)row * lds + col];
  dst[(long)row * ldd + col] = v > 0.f ? v : 0.f;
}

// out[c] = act( x[0:K] . W[:,c] + b[c] ), single block
__global__ __launch_bounds__(256)
void vecmat_kernel(const float* __restrict__ x, int K,
                   const float* __restrict__ W, int N,
                   const float* __restrict__ b, int act,
                   const float* __restrict__ pa, float* __restrict__ out)
{
  const float a = pa ? pa[0] : 0.f;
  for (int c = threadIdx.x; c < N; c += blockDim.x) {
    float s = b ? b[c] : 0.f;
    for (int k = 0; k < K; ++k) s += x[k] * W[(long)k * N + c];
    if (act == 2)      s = s >= 0.f ? s : a * s;
    else if (act == 3) s = sigf(s);
    out[c] = s;
  }
}

// out[i] = sigmoid( X[i,0:256] . w + b ), wave per row
__global__ __launch_bounds__(256)
void rowdot_sigmoid_kernel(const float* __restrict__ X, const float* __restrict__ w,
                           const float* __restrict__ b, int M, float* __restrict__ out)
{
  const int row = blockIdx.x * (blockDim.x >> 5) + (threadIdx.x >> 5);
  const int lane = threadIdx.x & 31;
  if (row >= M) return;
  const float* xr = X + (long)row * 256;
  float s = 0.f;
#pragma unroll
  for (int j = 0; j < 8; ++j) s += xr[lane + j*32] * w[lane + j*32];
  for (int off = 16; off > 0; off >>= 1) s += __shfl_down(s, off);
  if (lane == 0) out[row] = sigf(s + b[0]);
}

// fc3 -> softmax(action) -> argmax one-hot; builds p_in = [ftd(1024), onehot(10)]
__global__ __launch_bounds__(256)
void head_kernel(const float* __restrict__ act2, const float* __restrict__ fc3w,
                 const float* __restrict__ fc3b, const float* __restrict__ ftd,
                 float* __restrict__ action_out, float* __restrict__ pin)
{
  __shared__ float logits[16];
  __shared__ int amax;
  const int t = threadIdx.x;
  if (t < 10) {
    float s = fc3b[t];
    for (int k = 0; k < 256; ++k) s += act2[k] * fc3w[k * 10 + t];
    logits[t] = s;
  }
  __syncthreads();
  if (t == 0) {
    float mx = logits[0]; int mi = 0;
    for (int i = 1; i < 10; ++i) if (logits[i] > mx) { mx = logits[i]; mi = i; }
    float ex[10], sum = 0.f;
    for (int i = 0; i < 10; ++i) { ex[i] = __expf(logits[i] - mx); sum += ex[i]; }
    for (int i = 0; i < 10; ++i) action_out[i] = ex[i] / sum;
    amax = mi;
  }
  __syncthreads();
  for (int i = t; i < 1024; i += blockDim.x) pin[i] = ftd[i];
  if (t < 10) pin[1024 + t] = (t == amax) ? 1.f : 0.f;
}

__global__ void zero_kernel(float* p, int n)
{ int i = blockIdx.x * blockDim.x + threadIdx.x; if (i < n) p[i] = 0.f; }

__global__ void add_scalar_kernel(float* p, const float* s, int n)
{ int i = blockIdx.x * blockDim.x + threadIdx.x; if (i < n) p[i] += s[0]; }

// ---------------------------------------------------------------------------
static void launch_gemm(hipStream_t s, int act, int mode,
                        const float* A, int M, int K, int lda,
                        const float* W, int N, const float* bias,
                        const float* rowScalar, const float* extraVec,
                        const float* rowW, float* out, int ldc, const float* pa)
{
  dim3 g((N + 63) / 64, (M + 63) / 64), b(128);
  if (mode == 1)
    gemm_wmma_kernel<0,1><<<g, b, 0, s>>>(A, M, K, lda, W, N, bias,
                                          rowScalar, extraVec, rowW, out, ldc, pa);
  else if (act == 2)
    gemm_wmma_kernel<2,0><<<g, b, 0, s>>>(A, M, K, lda, W, N, bias,
                                          rowScalar, extraVec, rowW, out, ldc, pa);
  else
    gemm_wmma_kernel<0,0><<<g, b, 0, s>>>(A, M, K, lda, W, N, bias,
                                          rowScalar, extraVec, rowW, out, ldc, pa);
}

extern "C" void kernel_launch(void* const* d_in, const int* in_sizes, int n_in,
                              void* d_out_v, int out_size, void* d_ws, size_t ws_size,
                              hipStream_t stream)
{
  (void)in_sizes; (void)n_in; (void)out_size; (void)ws_size;
  const float* feat[2] = { (const float*)d_in[I_FEAT_G], (const float*)d_in[I_FEAT_GOAL] };
  const int*   srcs[2] = { (const int*)d_in[I_SRC_G],    (const int*)d_in[I_SRC_GOAL] };
  const int*   dsts[2] = { (const int*)d_in[I_DST_G],    (const int*)d_in[I_DST_GOAL] };
  const float* pa = (const float*)d_in[I_PRELU_A];
  float* out = (float*)d_out_v;

  // workspace layout (~360 MB of f32)
  float* ws = (float*)d_ws;
  float* BIG[2] = { ws, ws + (size_t)N_OBJ * 512 };     // [N,512] concat buffers
  float* BUFA = BIG[1] + (size_t)N_OBJ * 512;           // [N,256]
  float* BUFB = BUFA + (size_t)N_OBJ * 256;
  float* BUFC = BUFB + (size_t)N_OBJ * 256;
  float* GS   = BUFC + (size_t)N_OBJ * 256;             // [N,4]
  float* GD   = GS   + (size_t)N_OBJ * 4;
  float* FTD  = GD   + (size_t)N_OBJ * 4;               // [1024]
  float* PIN  = FTD + 1024;                             // [1040]
  float* CV1  = PIN + 1040;                             // [256]
  float* CV2  = CV1 + 256;
  float* A1   = CV2 + 256;
  float* A2v  = A1 + 256;
  float* QS1  = A2v + 256;
  float* QS2  = QS1 + 256;

  struct LayerP { const float *W, *Wself, *b, *wgd, *wgs; int fi; };
  const LayerP L[2] = {
    { (const float*)d_in[I_L0_W], (const float*)d_in[I_L0_WSELF], (const float*)d_in[I_L0_B],
      (const float*)d_in[I_L0_WGD], (const float*)d_in[I_L0_WGS], 300 },
    { (const float*)d_in[I_L1_W], (const float*)d_in[I_L1_WSELF], (const float*)d_in[I_L1_B],
      (const float*)d_in[I_L1_WGD], (const float*)d_in[I_L1_WGS], 256 },
  };

  const int gateBlocks = (N_OBJ + 7) / 8;
  const int edgeBlocks = (E_PER + 7) / 8;
  const int reluBlocks = (N_OBJ * 256 + 255) / 256;

  // ---- two GNN branches ----
  for (int b = 0; b < 2; ++b) {
    const float* h = feat[b];
    float* aggBuf[2] = { BUFB, BUFA };
    for (int l = 0; l < 2; ++l) {
      const LayerP& lp = L[l];
      float* AGG = aggBuf[l];
      gates_kernel<<<gateBlocks, 256, 0, stream>>>(h, lp.fi, lp.wgs, lp.wgd, GS, GD, N_OBJ);
      // agg = h @ W_self + b
      launch_gemm(stream, 0, 0, h, N_OBJ, lp.fi, lp.fi,
                  lp.Wself, 256, lp.b, nullptr, nullptr, nullptr, AGG, 256, nullptr);
      for (int e = 0; e < 3; ++e) {
        launch_gemm(stream, 0, 0, h, N_OBJ, lp.fi, lp.fi,
                    lp.W + (size_t)e * lp.fi * 256, 256, nullptr,
                    nullptr, nullptr, nullptr, BUFC, 256, nullptr);
        scatter_kernel<<<edgeBlocks, 256, 0, stream>>>(BUFC, GS, GD, e,
            srcs[b] + (size_t)e * E_PER, dsts[b] + (size_t)e * E_PER, E_PER, AGG);
      }
      if (l == 0) {
        relu_copy_kernel<<<reluBlocks, 256, 0, stream>>>(AGG, AGG, N_OBJ, 256, 256, 256);
        h = AGG;                                     // BUFB
      } else {
        relu_copy_kernel<<<reluBlocks, 256, 0, stream>>>(AGG, BIG[b], N_OBJ, 256, 256, 512);
      }
    }
    // metric path -> BIG[b][:,256:512]
    launch_gemm(stream, 2, 0, feat[b], N_OBJ, 300, 300,
                (const float*)d_in[I_M1_W], 256, (const float*)d_in[I_M1_B],
                nullptr, nullptr, nullptr, BUFC, 256, pa);
    launch_gemm(stream, 2, 0, BUFC, N_OBJ, 256, 256,
                (const float*)d_in[I_M2_W], 256, (const float*)d_in[I_M2_B],
                nullptr, nullptr, nullptr, BIG[b] + 256, 512, pa);
  }

  // ---- scene: X = |B1-B0| in place, then ftd[c] = sum_i (X@W1+b1)[i,c]*w2[i] (+b2)
  {
    const long nBig = (long)N_OBJ * 512;
    absdiff_kernel<<<(int)((nBig + 255) / 256), 256, 0, stream>>>(BIG[1], BIG[0], nBig);
  }
  zero_kernel<<<4, 256, 0, stream>>>(FTD, 1024);
  launch_gemm(stream, 0, 1, BIG[1], N_OBJ, 512, 512,
              (const float*)d_in[I_SC1_W], 1024, (const float*)d_in[I_SC1_B],
              nullptr, nullptr, (const float*)d_in[I_SC2_W], FTD, 0, nullptr);
  add_scalar_kernel<<<4, 256, 0, stream>>>(FTD, (const float*)d_in[I_SC2_B], 1024);

  // ---- action head ----
  vecmat_kernel<<<1, 256, 0, stream>>>(FTD, 1024, (const float*)d_in[I_FC1_W], 256,
              (const float*)d_in[I_FC1_B], 2, pa, A1);
  vecmat_kernel<<<1, 256, 0, stream>>>(A1, 256, (const float*)d_in[I_FC2_W], 256,
              (const float*)d_in[I_FC2_B], 2, pa, A2v);
  head_kernel<<<1, 256, 0, stream>>>(A2v, (const float*)d_in[I_FC3_W],
              (const float*)d_in[I_FC3_B], FTD, out, PIN);

  // ---- collapse the broadcast "rep" rows into per-column constants ----
  vecmat_kernel<<<1, 256, 0, stream>>>(PIN, 1034, (const float*)d_in[I_P1_W], 256,
              (const float*)d_in[I_P1_B], 0, nullptr, CV1);
  vecmat_kernel<<<1, 256, 0, stream>>>(PIN, 1034, (const float*)d_in[I_Q1O_W], 256,
              (const float*)d_in[I_Q1O_B], 0, nullptr, CV2);

  // ---- q-state path (1 row) ----
  vecmat_kernel<<<1, 256, 0, stream>>>(PIN, 1034, (const float*)d_in[I_Q1S_W], 256,
              (const float*)d_in[I_Q1S_B], 2, pa, QS1);
  vecmat_kernel<<<1, 256, 0, stream>>>(QS1, 256, (const float*)d_in[I_Q2S_W], 256,
              (const float*)d_in[I_Q2S_B], 2, pa, QS2);
  vecmat_kernel<<<1, 256, 0, stream>>>(QS2, 256, (const float*)d_in[I_Q3S_W], 4,
              (const float*)d_in[I_Q3S_B], 3, nullptr, out + 10 + 2 * N_OBJ);

  // ---- embedding ----
  launch_gemm(stream, 2, 0, (const float*)d_in[I_OBJECT_VEC], N_OBJ, 300, 300,
              (const float*)d_in[I_EMB1_W], 256, (const float*)d_in[I_EMB1_B],
              nullptr, nullptr, nullptr, BUFA, 256, pa);
  launch_gemm(stream, 2, 0, BUFA, N_OBJ, 256, 256,
              (const float*)d_in[I_EMB2_W], 256, (const float*)d_in[I_EMB2_B],
              nullptr, nullptr, nullptr, BUFB, 256, pa);        // E2 in BUFB

  // ---- p path: pred1 ----
  launch_gemm(stream, 2, 0, BUFB, N_OBJ, 256, 256,
              (const float*)d_in[I_P1_W] + (size_t)1034 * 256, 256, CV1,
              nullptr, nullptr, nullptr, BUFA, 256, pa);
  launch_gemm(stream, 2, 0, BUFA, N_OBJ, 256, 256,
              (const float*)d_in[I_P2_W], 256, (const float*)d_in[I_P2_B],
              nullptr, nullptr, nullptr, BUFC, 256, pa);
  rowdot_sigmoid_kernel<<<gateBlocks, 256, 0, stream>>>(BUFC,
              (const float*)d_in[I_P3_W], (const float*)d_in[I_P3_B], N_OBJ, out + 10);

  // ---- q-object path: pred2_obj (rank-1 term: pred1[row] * q1o_w[1290,:]) ----
  launch_gemm(stream, 2, 0, BUFB, N_OBJ, 256, 256,
              (const float*)d_in[I_Q1O_W] + (size_t)1034 * 256, 256, CV2,
              out + 10, (const float*)d_in[I_Q1O_W] + (size_t)1290 * 256,
              nullptr, BUFA, 256, pa);
  launch_gemm(stream, 2, 0, BUFA, N_OBJ, 256, 256,
              (const float*)d_in[I_Q2O_W], 256, (const float*)d_in[I_Q2O_B],
              nullptr, nullptr, nullptr, BUFC, 256, pa);
  rowdot_sigmoid_kernel<<<gateBlocks, 256, 0, stream>>>(BUFC,
              (const float*)d_in[I_Q3O_W], (const float*)d_in[I_Q3O_B], N_OBJ,
              out + 10 + N_OBJ);
}
